// Forecast_32435593019598
// MI455X (gfx1250) — matrix operand order repaired and checked by reading, not compiled, not run on hardware
//
#include <hip/hip_runtime.h>
#include <hip/hip_bf16.h>
#include <math.h>

// ---------------------------------------------------------------------------
// CDNA5 (gfx1250) TrajGRU forecaster, round 3.
// Implicit-GEMM convs on v_wmma_f32_16x16x32_f16, wave32.
//  - Weights pre-packed in per-lane WMMA A-fragment order (direct global load)
//  - 2 (Cout) x 4 (spatial) register blocking -> 8 WMMA per 32-K chunk
//  - Bilinear warp fused into the 1x1 "ret" GEMM B-staging
//  - MODE templated (conv / transpose-conv), partial-unrolled staging
// ---------------------------------------------------------------------------

typedef __attribute__((ext_vector_type(16))) _Float16 v16h;
typedef __attribute__((ext_vector_type(8)))  _Float16 v8h;
typedef __attribute__((ext_vector_type(8)))  float    v8f;

#define WAVE 32

// ---------------------------------------------------------------------------
// Pack fp32 weights into f16 A-fragment order:
//   wpf[((kc*nMt + mt)*32 + lane)*16 + j]
//   lane: g=lane>>4, r=lane&15 ; m = mt*16 + r
//   j<8  -> k = kc*32 + g*8 + j ;  j>=8 -> k = kc*32 + 16 + g*8 + (j-8)
//  conv   source: w[Co][Ci][KH][KW] ; deconv source: w[Ci][Co][KH][KW]
// ---------------------------------------------------------------------------
__global__ void pack_wfrag_kernel(const float* __restrict__ w, _Float16* __restrict__ wpf,
                                  int Co, int CoPad, int Ktot, int Kpad, int KHW, int deconv)
{
    int idx = blockIdx.x * 256 + threadIdx.x;
    int total = Kpad * CoPad;
    if (idx >= total) return;
    int j    = idx & 15;
    int lane = (idx >> 4) & 31;
    int t    = idx >> 9;
    int nMt  = CoPad >> 4;
    int mt   = t % nMt;
    int kc   = t / nMt;
    int g = lane >> 4, r = lane & 15;
    int m  = mt * 16 + r;
    int kj = (j < 8) ? (g * 8 + j) : (16 + g * 8 + (j - 8));
    int k  = kc * 32 + kj;
    float v = 0.f;
    if (m < Co && k < Ktot) {
        if (!deconv) {
            v = w[(size_t)m * Ktot + k];
        } else {
            int ci = k / KHW;
            int tt = k - ci * KHW;
            v = w[((size_t)ci * Co + m) * KHW + tt];
        }
    }
    wpf[idx] = (_Float16)v;
}

// B fragment from LDS row Bt[n][32]: lane l (g=l>>4,r=l&15) -> j<8: row[g*8 + j], j>=8: row[16 + g*8 + (j-8)]
__device__ __forceinline__ v16h load_b_frag(const _Float16* row, int g)
{
    v8h lo = *(const v8h*)(row + g * 8);
    v8h hi = *(const v8h*)(row + 16 + g * 8);
    return __builtin_shufflevector(lo, hi, 0,1,2,3,4,5,6,7,8,9,10,11,12,13,14,15);
}

#define WMMA_F16(A, B, C) \
    __builtin_amdgcn_wmma_f32_16x16x32_f16(false, (A), false, (B), (short)0, (C), false, false)

// ---------------------------------------------------------------------------
// Implicit-GEMM conv (MODE=0) / stride-2 transpose-conv gather (MODE=1).
// One wave -> 32(Cout) x 64(spatial) tile, 8 WMMA per 32-K chunk.
// ---------------------------------------------------------------------------
template <int MODE>
__global__ __launch_bounds__(WAVE)
void conv_wmma_kernel(const float* __restrict__ x, const _Float16* __restrict__ wpf,
                      const float* __restrict__ bias, float* __restrict__ out,
                      int Cin, int Hout, int Wout, int Cout,
                      int KH, int KW, int pad,
                      int Kpad, int CoPad, int accum, int act,
                      long outBS, long outOff)
{
    const int S      = Hout * Wout;
    const int b      = blockIdx.z;
    const int nMt    = CoPad >> 4;
    const int mtBase = blockIdx.y * 2;
    const int sBase  = blockIdx.x * 64;
    const int lane   = threadIdx.x;
    const int g      = lane >> 4;
    const int r      = lane & 15;
    const int Hin    = MODE ? (Hout >> 1) : Hout;
    const int Win    = MODE ? (Wout >> 1) : Wout;
    const int KHW    = KH * KW;
    const int Ktot   = Cin * KHW;
    const float* xb  = x + (size_t)b * Cin * Hin * Win;
    const v16h* wf   = (const v16h*)wpf;

    const int oh0 = sBase / Wout;
    const int ow0 = sBase - oh0 * Wout;

    __shared__ __attribute__((aligned(16))) _Float16 Bt[64][32];

    v8f acc[2][4] = {};

    const int nKc = Kpad >> 5;
    for (int kc = 0; kc < nKc; ++kc) {
        // ---- stage B tile [64 n][32 k]; lane owns one k slot
        int kk = kc * 32 + lane;
        if (kk < Ktot) {
            int ci  = kk / KHW;
            int rem = kk - ci * KHW;
            int kh  = rem / KW;
            int kw  = rem - kh * KW;
            const float* xc = xb + (size_t)ci * Hin * Win;
            if (MODE == 0) {
                int ih  = oh0 + kh - pad;
                int iw0 = kw - pad;
                int iw  = ow0 + iw0;
                int owc = ow0;
#pragma unroll 4
                for (int n = 0; n < 64; ++n) {
                    float v = 0.f;
                    if (ih >= 0 && ih < Hin && iw >= 0 && iw < Win)
                        v = xc[ih * Win + iw];
                    Bt[n][lane] = (_Float16)v;
                    ++iw;
                    if (++owc == Wout) { owc = 0; iw = iw0; ++ih; }
                }
            } else {
                int ohc = oh0, owc = ow0;
#pragma unroll 4
                for (int n = 0; n < 64; ++n) {
                    float v = 0.f;
                    int ih2 = ohc + 1 - kh;
                    int iw2 = owc + 1 - kw;
                    if (ih2 >= 0 && iw2 >= 0 && !(ih2 & 1) && !(iw2 & 1)) {
                        int ih = ih2 >> 1, iw = iw2 >> 1;
                        if (ih < Hin && iw < Win) v = xc[ih * Win + iw];
                    }
                    Bt[n][lane] = (_Float16)v;
                    if (++owc == Wout) { owc = 0; ++ohc; }
                }
            }
        } else {
#pragma unroll 8
            for (int n = 0; n < 64; ++n) Bt[n][lane] = (_Float16)0.f;
        }
        __syncthreads();
        // ---- A fragments: contiguous global loads (pre-packed WMMA order)
        v16h a0 = wf[((size_t)kc * nMt + mtBase)     * 32 + lane];
        v16h a1 = wf[((size_t)kc * nMt + mtBase + 1) * 32 + lane];
#pragma unroll
        for (int nt = 0; nt < 4; ++nt) {
            v16h bfv = load_b_frag(&Bt[nt * 16 + r][0], g);
            acc[0][nt] = WMMA_F16(a0, bfv, acc[0][nt]);
            acc[1][nt] = WMMA_F16(a1, bfv, acc[1][nt]);
        }
        __syncthreads();
    }

    // ---- epilogue: +bias, optional accumulate, optional leaky(0.2)
#pragma unroll
    for (int mt = 0; mt < 2; ++mt) {
#pragma unroll
        for (int nt = 0; nt < 4; ++nt) {
#pragma unroll
            for (int v = 0; v < 8; ++v) {
                int co = (mtBase + mt) * 16 + v + 8 * g;
                int s  = sBase + nt * 16 + r;
                if (co < Cout) {
                    size_t oi = (size_t)b * outBS + (size_t)co * S + s + outOff;
                    float val = acc[mt][nt][v] + bias[co];
                    if (accum) val += out[oi];
                    if (act == 1) val = val > 0.f ? val : 0.2f * val;
                    out[oi] = val;
                }
            }
        }
    }
}

// ---------------------------------------------------------------------------
// Fused warp + 1x1 "ret" GEMM: B tile produced by bilinear flow sampling of
// the hidden state (warped tensor never materialized).
// ---------------------------------------------------------------------------
__device__ __forceinline__ float samp(const float* img, int H, int W, int xi, int yi)
{
    if (xi < 0 || xi >= W || yi < 0 || yi >= H) return 0.f;
    return img[yi * W + xi];
}

__global__ __launch_bounds__(WAVE)
void retwarp_wmma_kernel(const float* __restrict__ hstate, const float* __restrict__ flows,
                         const _Float16* __restrict__ wpf, const float* __restrict__ bias,
                         float* __restrict__ out,
                         int C, int H, int W, int L, int Cout, int Kpad, int CoPad)
{
    const int S      = H * W;
    const int b      = blockIdx.z;
    const int nMt    = CoPad >> 4;
    const int mtBase = blockIdx.y * 2;
    const int sBase  = blockIdx.x * 64;
    const int lane   = threadIdx.x;
    const int g      = lane >> 4;
    const int r      = lane & 15;
    const int Ktot   = L * C;
    const float* hb  = hstate + (size_t)b * C * S;
    const float* flb = flows  + (size_t)b * 2 * L * S;
    const v16h* wf   = (const v16h*)wpf;

    const int yy0 = sBase / W;
    const int xx0 = sBase - yy0 * W;

    __shared__ __attribute__((aligned(16))) _Float16 Bt[64][32];

    v8f acc[2][4] = {};

    const int nKc = Kpad >> 5;
    for (int kc = 0; kc < nKc; ++kc) {
        int kk = kc * 32 + lane;
        if (kk < Ktot) {
            int fl = kk / C;
            int c  = kk - fl * C;
            const float* img = hb + (size_t)c * S;
            const float* fx  = flb + (size_t)(2 * fl) * S;
            const float* fy  = flb + (size_t)(2 * fl + 1) * S;
            int s  = sBase;
            int yy = yy0, xx = xx0;
#pragma unroll 4
            for (int n = 0; n < 64; ++n) {
                float px = (float)xx - fx[s];
                float py = (float)yy - fy[s];
                float x0f = floorf(px), y0f = floorf(py);
                float dx = px - x0f, dy = py - y0f;
                int x0 = (int)x0f, y0 = (int)y0f;
                float v00 = samp(img, H, W, x0,     y0);
                float v01 = samp(img, H, W, x0,     y0 + 1);
                float v10 = samp(img, H, W, x0 + 1, y0);
                float v11 = samp(img, H, W, x0 + 1, y0 + 1);
                float val = v00 * (1.f - dx) * (1.f - dy)
                          + v01 * (1.f - dx) * dy
                          + v10 * dx * (1.f - dy)
                          + v11 * dx * dy;
                Bt[n][lane] = (_Float16)val;
                ++s;
                if (++xx == W) { xx = 0; ++yy; }
            }
        } else {
#pragma unroll 8
            for (int n = 0; n < 64; ++n) Bt[n][lane] = (_Float16)0.f;
        }
        __syncthreads();
        v16h a0 = wf[((size_t)kc * nMt + mtBase)     * 32 + lane];
        v16h a1 = wf[((size_t)kc * nMt + mtBase + 1) * 32 + lane];
#pragma unroll
        for (int nt = 0; nt < 4; ++nt) {
            v16h bfv = load_b_frag(&Bt[nt * 16 + r][0], g);
            acc[0][nt] = WMMA_F16(a0, bfv, acc[0][nt]);
            acc[1][nt] = WMMA_F16(a1, bfv, acc[1][nt]);
        }
        __syncthreads();
    }

#pragma unroll
    for (int mt = 0; mt < 2; ++mt) {
#pragma unroll
        for (int nt = 0; nt < 4; ++nt) {
#pragma unroll
            for (int v = 0; v < 8; ++v) {
                int co = (mtBase + mt) * 16 + v + 8 * g;
                int s  = sBase + nt * 16 + r;
                if (co < Cout) {
                    size_t oi = ((size_t)b * Cout + co) * S + s;
                    out[oi] = acc[mt][nt][v] + bias[co];
                }
            }
        }
    }
}

// ---------------------------------------------------------------------------
// Fused TrajGRU gate update (in place on h).
// ---------------------------------------------------------------------------
__global__ void gates_kernel(const float* __restrict__ g, const float* __restrict__ gi,
                             float* __restrict__ h, int B, int ch, int S, int hasInput)
{
    int idx = blockIdx.x * 256 + threadIdx.x;
    int total = B * ch * S;
    if (idx >= total) return;
    int s = idx % S;
    int t = idx / S;
    int c = t % ch;
    int b = t / ch;
    size_t base = (size_t)b * 3 * ch * S;
    float hz = g[base + (size_t)c * S + s];
    float hr = g[base + (size_t)(c + ch) * S + s];
    float hh = g[base + (size_t)(c + 2 * ch) * S + s];
    float z, rr, pre;
    if (hasInput) {
        float iz = gi[base + (size_t)c * S + s];
        float ir = gi[base + (size_t)(c + ch) * S + s];
        float ih = gi[base + (size_t)(c + 2 * ch) * S + s];
        z  = 1.f / (1.f + __expf(-(iz + hz)));
        rr = 1.f / (1.f + __expf(-(ir + hr)));
        pre = ih + rr * hh;
    } else {
        z  = 1.f / (1.f + __expf(-hz));
        rr = 1.f / (1.f + __expf(-hr));
        pre = rr * hh;
    }
    float nn = pre > 0.f ? pre : 0.2f * pre;
    size_t hidx = ((size_t)b * ch + c) * S + s;
    float hv = h[hidx];
    h[hidx] = z * hv + (1.f - z) * nn;
}

// ---------------------------------------------------------------------------
// Host orchestration
// ---------------------------------------------------------------------------
static inline int rup(int v, int m) { return (v + m - 1) / m * m; }

extern "C" void kernel_launch(void* const* d_in, const int* in_sizes, int n_in,
                              void* d_out, int out_size, void* d_ws, size_t ws_size,
                              hipStream_t stream)
{
    const int B = 4;
    const float* h1_in = (const float*)d_in[0];   // [4,96,16,16]
    const float* h2_in = (const float*)d_in[1];   // [4,96,32,32]
    const float* h3_in = (const float*)d_in[2];   // [4,64,64,64]
    int p = 3;
    const float* w_l1_h2f = (const float*)d_in[p++]; const float* b_l1_h2f = (const float*)d_in[p++];
    const float* w_l1_flw = (const float*)d_in[p++]; const float* b_l1_flw = (const float*)d_in[p++];
    const float* w_l1_ret = (const float*)d_in[p++]; const float* b_l1_ret = (const float*)d_in[p++];
    const float* w_l2_h2f = (const float*)d_in[p++]; const float* b_l2_h2f = (const float*)d_in[p++];
    const float* w_l2_flw = (const float*)d_in[p++]; const float* b_l2_flw = (const float*)d_in[p++];
    const float* w_l2_ret = (const float*)d_in[p++]; const float* b_l2_ret = (const float*)d_in[p++];
    const float* w_l2_i2f = (const float*)d_in[p++]; const float* b_l2_i2f = (const float*)d_in[p++];
    const float* w_l2_i2h = (const float*)d_in[p++]; const float* b_l2_i2h = (const float*)d_in[p++];
    const float* w_l3_h2f = (const float*)d_in[p++]; const float* b_l3_h2f = (const float*)d_in[p++];
    const float* w_l3_flw = (const float*)d_in[p++]; const float* b_l3_flw = (const float*)d_in[p++];
    const float* w_l3_ret = (const float*)d_in[p++]; const float* b_l3_ret = (const float*)d_in[p++];
    const float* w_l3_i2f = (const float*)d_in[p++]; const float* b_l3_i2f = (const float*)d_in[p++];
    const float* w_l3_i2h = (const float*)d_in[p++]; const float* b_l3_i2h = (const float*)d_in[p++];
    const float* w_dc1 = (const float*)d_in[p++]; const float* b_dc1 = (const float*)d_in[p++];
    const float* w_dc2 = (const float*)d_in[p++]; const float* b_dc2 = (const float*)d_in[p++];
    const float* w_dc3 = (const float*)d_in[p++]; const float* b_dc3 = (const float*)d_in[p++];
    const float* w_c1  = (const float*)d_in[p++]; const float* b_c1  = (const float*)d_in[p++];
    const float* w_c2  = (const float*)d_in[p++]; const float* b_c2  = (const float*)d_in[p++];

    const int out_len = out_size / (B * 128 * 128);   // [B, T, 1, 128, 128]

    // ---- workspace carve-up
    char* ws = (char*)d_ws;
    size_t cur = 0;
    auto alloc = [&](size_t bytes) -> void* {
        void* ptr = ws + cur;
        cur += (bytes + 255) & ~(size_t)255;
        return ptr;
    };
    float* h1b  = (float*)alloc((size_t)B * 96 * 256  * 4);
    float* h2b  = (float*)alloc((size_t)B * 96 * 1024 * 4);
    float* h3b  = (float*)alloc((size_t)B * 64 * 4096 * 4);
    float* fbuf = (float*)alloc((size_t)B * 32 * 4096 * 4);
    float* flwb = (float*)alloc((size_t)B * 32 * 4096 * 4);
    float* gbuf = (float*)alloc((size_t)B * 192 * 4096 * 4);
    float* gib  = (float*)alloc((size_t)B * 192 * 4096 * 4);
    float* dbuf = (float*)alloc((size_t)B * 96 * 4096 * 4);
    float* cbuf = (float*)alloc((size_t)B * 8 * 16384 * 4);

    // ---- pack all weights into f16 WMMA A-fragment order, once per call
    struct PW { _Float16* wp; int Kpad, CoPad; };
    auto pack = [&](const float* w, int Co, int Cin, int KH, int KW, int deconv) -> PW {
        int KHW = KH * KW, Ktot = Cin * KHW;
        int Kpad = rup(Ktot, 32), CoPad = rup(Co, 32);
        _Float16* wp = (_Float16*)alloc((size_t)Kpad * CoPad * sizeof(_Float16));
        int total = Kpad * CoPad;
        pack_wfrag_kernel<<<(total + 255) / 256, 256, 0, stream>>>(w, wp, Co, CoPad, Ktot, Kpad, KHW, deconv);
        return PW{wp, Kpad, CoPad};
    };
    PW p_l1_h2f = pack(w_l1_h2f, 32, 96, 5, 5, 0);
    PW p_l1_flw = pack(w_l1_flw, 26, 32, 5, 5, 0);
    PW p_l1_ret = pack(w_l1_ret, 288, 1248, 1, 1, 0);
    PW p_l2_h2f = pack(w_l2_h2f, 32, 96, 5, 5, 0);
    PW p_l2_flw = pack(w_l2_flw, 26, 32, 5, 5, 0);
    PW p_l2_ret = pack(w_l2_ret, 288, 1248, 1, 1, 0);
    PW p_l2_i2f = pack(w_l2_i2f, 32, 96, 5, 5, 0);
    PW p_l2_i2h = pack(w_l2_i2h, 288, 96, 5, 5, 0);
    PW p_l3_h2f = pack(w_l3_h2f, 32, 64, 5, 5, 0);
    PW p_l3_flw = pack(w_l3_flw, 18, 32, 5, 5, 0);
    PW p_l3_ret = pack(w_l3_ret, 192, 576, 1, 1, 0);
    PW p_l3_i2f = pack(w_l3_i2f, 32, 96, 5, 5, 0);
    PW p_l3_i2h = pack(w_l3_i2h, 192, 96, 3, 3, 0);
    PW p_dc1 = pack(w_dc1, 96, 96, 3, 3, 1);
    PW p_dc2 = pack(w_dc2, 96, 96, 3, 3, 1);
    PW p_dc3 = pack(w_dc3,  8, 64, 3, 3, 1);
    PW p_c1  = pack(w_c1,   8,  8, 3, 3, 0);
    PW p_c2  = pack(w_c2,   1,  8, 1, 1, 0);

    // ---- copy hidden states (updated in place per step)
    hipMemcpyAsync(h1b, h1_in, (size_t)B * 96 * 256  * 4, hipMemcpyDeviceToDevice, stream);
    hipMemcpyAsync(h2b, h2_in, (size_t)B * 96 * 1024 * 4, hipMemcpyDeviceToDevice, stream);
    hipMemcpyAsync(h3b, h3_in, (size_t)B * 64 * 4096 * 4, hipMemcpyDeviceToDevice, stream);

    auto conv = [&](const float* x, PW pw, const float* bias, float* out,
                    int Cin, int Hout, int Wout, int Cout, int KH, int KW,
                    int pad, int mode, int accum, int act, long outBS, long outOff) {
        dim3 grid((Hout * Wout) / 64, pw.CoPad / 32, B);
        if (mode == 0)
            conv_wmma_kernel<0><<<grid, WAVE, 0, stream>>>(x, pw.wp, bias, out,
                Cin, Hout, Wout, Cout, KH, KW, pad, pw.Kpad, pw.CoPad, accum, act, outBS, outOff);
        else
            conv_wmma_kernel<1><<<grid, WAVE, 0, stream>>>(x, pw.wp, bias, out,
                Cin, Hout, Wout, Cout, KH, KW, pad, pw.Kpad, pw.CoPad, accum, act, outBS, outOff);
    };
    auto retwarp = [&](const float* h, const float* fl, PW pw, const float* bias,
                       float* out, int C, int H, int W, int L, int Cout) {
        dim3 grid((H * W) / 64, pw.CoPad / 32, B);
        retwarp_wmma_kernel<<<grid, WAVE, 0, stream>>>(h, fl, pw.wp, bias, out,
            C, H, W, L, Cout, pw.Kpad, pw.CoPad);
    };
    auto gates = [&](const float* g, const float* gi, float* h, int ch, int S, int hasX) {
        int total = B * ch * S;
        gates_kernel<<<(total + 255) / 256, 256, 0, stream>>>(g, gi, h, B, ch, S, hasX);
    };

    for (int t = 0; t < out_len; ++t) {
        // ---------- layer 1 cell (ch=96, 16x16, L=13, x=None) ----------
        conv(h1b, p_l1_h2f, b_l1_h2f, fbuf, 96, 16, 16, 32, 5, 5, 2, 0, 0, 1, 32L * 256, 0);
        conv(fbuf, p_l1_flw, b_l1_flw, flwb, 32, 16, 16, 26, 5, 5, 2, 0, 0, 0, 26L * 256, 0);
        retwarp(h1b, flwb, p_l1_ret, b_l1_ret, gbuf, 96, 16, 16, 13, 288);
        gates(gbuf, nullptr, h1b, 96, 256, 0);
        conv(h1b, p_dc1, b_dc1, dbuf, 96, 32, 32, 96, 3, 3, 1, 1, 0, 0, 96L * 1024, 0);

        // ---------- layer 2 cell (ch=96, 32x32, L=13, x=dbuf) ----------
        conv(h2b, p_l2_h2f, b_l2_h2f, fbuf, 96, 32, 32, 32, 5, 5, 2, 0, 0, 0, 32L * 1024, 0);
        conv(dbuf, p_l2_i2f, b_l2_i2f, fbuf, 96, 32, 32, 32, 5, 5, 2, 0, 1, 1, 32L * 1024, 0);
        conv(fbuf, p_l2_flw, b_l2_flw, flwb, 32, 32, 32, 26, 5, 5, 2, 0, 0, 0, 26L * 1024, 0);
        retwarp(h2b, flwb, p_l2_ret, b_l2_ret, gbuf, 96, 32, 32, 13, 288);
        conv(dbuf, p_l2_i2h, b_l2_i2h, gib, 96, 32, 32, 288, 5, 5, 2, 0, 0, 0, 288L * 1024, 0);
        gates(gbuf, gib, h2b, 96, 1024, 1);
        conv(h2b, p_dc2, b_dc2, dbuf, 96, 64, 64, 96, 3, 3, 1, 1, 0, 0, 96L * 4096, 0);

        // ---------- layer 3 cell (ch=64, 64x64, L=9, k=3, x=dbuf) ----------
        conv(h3b, p_l3_h2f, b_l3_h2f, fbuf, 64, 64, 64, 32, 5, 5, 2, 0, 0, 0, 32L * 4096, 0);
        conv(dbuf, p_l3_i2f, b_l3_i2f, fbuf, 96, 64, 64, 32, 5, 5, 2, 0, 1, 1, 32L * 4096, 0);
        conv(fbuf, p_l3_flw, b_l3_flw, flwb, 32, 64, 64, 18, 5, 5, 2, 0, 0, 0, 18L * 4096, 0);
        retwarp(h3b, flwb, p_l3_ret, b_l3_ret, gbuf, 64, 64, 64, 9, 192);
        conv(dbuf, p_l3_i2h, b_l3_i2h, gib, 96, 64, 64, 192, 3, 3, 1, 0, 0, 0, 192L * 4096, 0);
        gates(gbuf, gib, h3b, 64, 4096, 1);
        conv(h3b, p_dc3, b_dc3, dbuf, 64, 128, 128, 8, 3, 3, 1, 1, 0, 0, 8L * 16384, 0);

        // ---------- output head ----------
        conv(dbuf, p_c1, b_c1, cbuf, 8, 128, 128, 8, 3, 3, 1, 0, 0, 0, 8L * 16384, 0);
        conv(cbuf, p_c2, b_c2, (float*)d_out, 8, 128, 128, 1, 1, 1, 0, 0, 0, 0,
             (long)out_len * 16384, (long)t * 16384);
    }
    (void)in_sizes; (void)n_in; (void)ws_size;
}
